// MiniGPT_42013370090226
// MI455X (gfx1250) — compile-verified, matrix operand order checked by reading, not run on hardware
//
#include <hip/hip_runtime.h>
#include <hip/hip_bf16.h>

typedef _Float16 h4   __attribute__((ext_vector_type(4)));
typedef _Float16 h8   __attribute__((ext_vector_type(8)));
typedef _Float16 v16h __attribute__((ext_vector_type(16)));
typedef float    v8f  __attribute__((ext_vector_type(8)));

#define LDS_STRIDE 40   // halves; 80 B per row -> rows stay 16B aligned

// ---------------------------------------------------------------------------
// Embedding: x[n,c] = wte[idx[n],c] + wpe[n%T,c]
// ---------------------------------------------------------------------------
__global__ __launch_bounds__(256)
void embed_k(const int* __restrict__ idx, const float* __restrict__ wte,
             const float* __restrict__ wpe, float* __restrict__ x,
             int nrows, int C, int T)
{
    long i = (long)blockIdx.x * 256 + threadIdx.x;
    long tot = (long)nrows * C;
    if (i >= tot) return;
    int n = (int)(i / C), c = (int)(i % C);
    int t = n % T;
    x[i] = wte[(long)idx[n] * C + c] + wpe[(long)t * C + c];
}

// ---------------------------------------------------------------------------
// LayerNorm over rows of length C (one block of 256 threads per row)
// ---------------------------------------------------------------------------
__global__ __launch_bounds__(256)
void layernorm_k(const float* __restrict__ x, const float* __restrict__ w,
                 const float* __restrict__ b, float* __restrict__ out, int C)
{
    __shared__ float red[256];
    const int row = blockIdx.x;
    const float* xr = x + (long)row * C;

    float s = 0.f;
    for (int c = threadIdx.x; c < C; c += 256) s += xr[c];
    red[threadIdx.x] = s;
    __syncthreads();
    for (int off = 128; off > 0; off >>= 1) {
        if (threadIdx.x < off) red[threadIdx.x] += red[threadIdx.x + off];
        __syncthreads();
    }
    const float mean = red[0] / (float)C;
    __syncthreads();

    float s2 = 0.f;
    for (int c = threadIdx.x; c < C; c += 256) {
        float d = xr[c] - mean;
        s2 += d * d;
    }
    red[threadIdx.x] = s2;
    __syncthreads();
    for (int off = 128; off > 0; off >>= 1) {
        if (threadIdx.x < off) red[threadIdx.x] += red[threadIdx.x + off];
        __syncthreads();
    }
    const float rstd = rsqrtf(red[0] / (float)C + 1e-5f);

    for (int c = threadIdx.x; c < C; c += 256)
        out[(long)row * C + c] = (xr[c] - mean) * rstd * w[c] + b[c];
}

// Build a 16-half fragment from one f32 row: halves 0..7 <- p[0..7],
// halves 8..15 <- p[16..23]  (CDNA5 16-bit fragment K pattern, kb pre-added).
__device__ __forceinline__ v16h load_frag_f32row(const float* p, float sc)
{
    float4 f0 = *(const float4*)(p);
    float4 f1 = *(const float4*)(p + 4);
    float4 f2 = *(const float4*)(p + 16);
    float4 f3 = *(const float4*)(p + 20);
    v16h r;
    r[0]  = (_Float16)(f0.x * sc); r[1]  = (_Float16)(f0.y * sc);
    r[2]  = (_Float16)(f0.z * sc); r[3]  = (_Float16)(f0.w * sc);
    r[4]  = (_Float16)(f1.x * sc); r[5]  = (_Float16)(f1.y * sc);
    r[6]  = (_Float16)(f1.z * sc); r[7]  = (_Float16)(f1.w * sc);
    r[8]  = (_Float16)(f2.x * sc); r[9]  = (_Float16)(f2.y * sc);
    r[10] = (_Float16)(f2.z * sc); r[11] = (_Float16)(f2.w * sc);
    r[12] = (_Float16)(f3.x * sc); r[13] = (_Float16)(f3.y * sc);
    r[14] = (_Float16)(f3.z * sc); r[15] = (_Float16)(f3.w * sc);
    return r;
}

// ---------------------------------------------------------------------------
// WMMA GEMM: out[M,N] = A[M,K] @ W + (bias) (+gelu) (+res), f32 in/out,
// f16 fragments, f32 accumulate (v_wmma_f32_16x16x32_f16).
//   !B_TRANS: W is [K,N] row-major; N must be a multiple of 256.
//    B_TRANS: W is [N,K] row-major (tied lm_head, wte [V,C]); N ragged, guarded.
// Block: 256 threads = 8 wave32s, tile 64(M) x 256(N), K-step 32.
// Waves 2(M) x 4(N); each wave computes 32x64 = 2x4 WMMA tiles.
// Software-pipelined, double-buffered LDS, one barrier per K-chunk.
// ---------------------------------------------------------------------------
template<bool HAS_BIAS, bool DO_GELU, bool DO_RES, bool B_TRANS>
__global__ __launch_bounds__(256)
void gemm_wmma(const float* __restrict__ A, const float* __restrict__ W,
               const float* __restrict__ bias, const float* __restrict__ res,
               float* __restrict__ out, int M, int N, int K)
{
    __shared__ __align__(16) _Float16 As[2][64  * LDS_STRIDE]; // [m][k]
    __shared__ __align__(16) _Float16 Bs[2][256 * LDS_STRIDE]; // [n][k]

    const int tid  = threadIdx.x;
    const int lane = tid & 31;
    const int wave = tid >> 5;      // 0..7
    const int wm   = wave >> 2;     // 0..1
    const int wn   = wave & 3;      // 0..3
    const int m0   = blockIdx.y * 64;
    const int n0   = blockIdx.x * 256;

    v8f acc[2][4] = {};
    const int kb   = (lane >> 4) << 3;
    const int rsel = lane & 15;
    const int nChunks = K >> 5;

    float4 aReg[2];
    float4 bReg[8];

    auto loadRegs = [&](int k0) {
        #pragma unroll
        for (int i = 0; i < 2; ++i) {
            int e4 = tid + i * 256;
            int r = e4 >> 3, c4 = e4 & 7;
            aReg[i] = *(const float4*)&A[(long)(m0 + r) * K + k0 + c4 * 4];
        }
        #pragma unroll
        for (int i = 0; i < 8; ++i) {
            int e4 = tid + i * 256;
            if (B_TRANS) {
                int nn = e4 >> 3, k4 = e4 & 7;
                int n = n0 + nn;
                if (n < N)
                    bReg[i] = *(const float4*)&W[(long)n * K + k0 + k4 * 4];
                else
                    bReg[i] = make_float4(0.f, 0.f, 0.f, 0.f);
            } else {
                int kk = e4 >> 6, n4 = e4 & 63;
                bReg[i] = *(const float4*)&W[(long)(k0 + kk) * N + n0 + n4 * 4];
            }
        }
    };

    auto storeLDS = [&](int buf) {
        #pragma unroll
        for (int i = 0; i < 2; ++i) {
            int e4 = tid + i * 256;
            int r = e4 >> 3, c4 = e4 & 7;
            h4 hv = { (_Float16)aReg[i].x, (_Float16)aReg[i].y,
                      (_Float16)aReg[i].z, (_Float16)aReg[i].w };
            *(h4*)&As[buf][r * LDS_STRIDE + c4 * 4] = hv;
        }
        #pragma unroll
        for (int i = 0; i < 8; ++i) {
            int e4 = tid + i * 256;
            if (B_TRANS) {
                int nn = e4 >> 3, k4 = e4 & 7;
                h4 hv = { (_Float16)bReg[i].x, (_Float16)bReg[i].y,
                          (_Float16)bReg[i].z, (_Float16)bReg[i].w };
                *(h4*)&Bs[buf][nn * LDS_STRIDE + k4 * 4] = hv;
            } else {
                int kk = e4 >> 6, n4 = e4 & 63;
                Bs[buf][(n4 * 4 + 0) * LDS_STRIDE + kk] = (_Float16)bReg[i].x;
                Bs[buf][(n4 * 4 + 1) * LDS_STRIDE + kk] = (_Float16)bReg[i].y;
                Bs[buf][(n4 * 4 + 2) * LDS_STRIDE + kk] = (_Float16)bReg[i].z;
                Bs[buf][(n4 * 4 + 3) * LDS_STRIDE + kk] = (_Float16)bReg[i].w;
            }
        }
    };

    loadRegs(0);
    storeLDS(0);
    __syncthreads();

    for (int c = 0; c < nChunks; ++c) {
        const int buf = c & 1;
        const bool more = (c + 1 < nChunks);
        if (more) loadRegs((c + 1) << 5);

        v16h af[2];
        #pragma unroll
        for (int mi = 0; mi < 2; ++mi) {
            const int arow = wm * 32 + mi * 16 + rsel;
            h8 alo = *(const h8*)&As[buf][arow * LDS_STRIDE + kb];
            h8 ahi = *(const h8*)&As[buf][arow * LDS_STRIDE + kb + 16];
            af[mi] = __builtin_shufflevector(alo, ahi,
                        0,1,2,3,4,5,6,7,8,9,10,11,12,13,14,15);
        }
        #pragma unroll
        for (int ni = 0; ni < 4; ++ni) {
            const int bcol = wn * 64 + ni * 16 + rsel;
            h8 blo = *(const h8*)&Bs[buf][bcol * LDS_STRIDE + kb];
            h8 bhi = *(const h8*)&Bs[buf][bcol * LDS_STRIDE + kb + 16];
            v16h bf = __builtin_shufflevector(blo, bhi,
                        0,1,2,3,4,5,6,7,8,9,10,11,12,13,14,15);
            #pragma unroll
            for (int mi = 0; mi < 2; ++mi) {
                acc[mi][ni] = __builtin_amdgcn_wmma_f32_16x16x32_f16(
                    false, af[mi], false, bf,
                    (short)0, acc[mi][ni], false, false);
            }
        }

        if (more) {
            storeLDS(buf ^ 1);
            __syncthreads();
        }
    }

    #pragma unroll
    for (int mi = 0; mi < 2; ++mi) {
        #pragma unroll
        for (int ni = 0; ni < 4; ++ni) {
            const int mbase = m0 + wm * 32 + mi * 16 + ((lane >> 4) << 3);
            const int ncol  = n0 + wn * 64 + ni * 16 + (lane & 15);
            if (ncol < N) {
                const float bv = HAS_BIAS ? bias[ncol] : 0.f;
                #pragma unroll
                for (int r = 0; r < 8; ++r) {
                    const int mrow = mbase + r;
                    float v = acc[mi][ni][r] + bv;
                    if (DO_GELU)
                        v = 0.5f * v * (1.0f + erff(v * 0.70710678118654752f));
                    if (DO_RES)
                        v += res[(long)mrow * N + ncol];
                    out[(long)mrow * N + ncol] = v;
                }
            }
        }
    }
}

// ---------------------------------------------------------------------------
// Flash attention with WMMA. One wave per 16 query rows; 4 independent waves
// per block (no block barriers -> per-wave causal trip counts). HD == 64.
// S = (Q*scale) @ K^T per 32-key tile -> online softmax in C layout ->
// P via per-wave LDS (C->A layout) -> O += P @ V.  qkv row: [q|k|v], stride 3C.
// ---------------------------------------------------------------------------
__global__ __launch_bounds__(128)
void attn_wmma_k(const float* __restrict__ qkv, float* __restrict__ y,
                 int Bn, int Hn, int T)
{
    __shared__ __align__(16) _Float16 Ptile[4][16 * LDS_STRIDE];

    const int lane = threadIdx.x & 31;
    const int wv   = threadIdx.x >> 5;
    const int g    = blockIdx.x * 4 + wv;   // global wave id
    const int QT   = T >> 4;                // q tiles of 16
    const int qt   = g % QT;
    const int hq   = (g / QT) % Hn;
    const int b    = g / (QT * Hn);
    const int C    = Hn * 64;
    const int C3   = 3 * C;
    const int qbase = qt << 4;

    const int kb   = (lane >> 4) << 3;      // 0 or 8
    const int rsel = lane & 15;
    _Float16* Pt = Ptile[wv];

    // persistent Q fragments (A layout), scale 1/8 folded in (exact in f16)
    const float* qrow = qkv + (long)(b * T + qbase + rsel) * C3 + hq * 64;
    v16h qf[2];
    #pragma unroll
    for (int kc = 0; kc < 2; ++kc)
        qf[kc] = load_frag_f32row(qrow + kc * 32 + kb, 0.125f);

    v8f oacc[4] = {};
    float m[8], l[8];
    #pragma unroll
    for (int r = 0; r < 8; ++r) { m[r] = -3.0e38f; l[r] = 0.f; }

    const int rowoff = qbase + ((lane >> 4) << 3);  // global row of element r=0

    for (int j0 = 0; j0 <= qbase; j0 += 32) {
        // ---- S = Q @ K^T for 32 keys (2 groups of 16)
        v8f sacc[2] = {};
        #pragma unroll
        for (int grp = 0; grp < 2; ++grp) {
            const float* krow = qkv + (long)(b * T + j0 + grp * 16 + rsel) * C3
                                + C + hq * 64;
            #pragma unroll
            for (int kc = 0; kc < 2; ++kc) {
                v16h kf = load_frag_f32row(krow + kc * 32 + kb, 1.0f);
                sacc[grp] = __builtin_amdgcn_wmma_f32_16x16x32_f16(
                    false, qf[kc], false, kf, (short)0, sacc[grp], false, false);
            }
        }

        // ---- causal mask + online softmax (row = rowoff+r, key = j0+grp*16+rsel)
        const int key0 = j0 + rsel;
        #pragma unroll
        for (int r = 0; r < 8; ++r) {
            const int row = rowoff + r;
            float s0 = (key0      <= row) ? sacc[0][r] : -1.0e30f;
            float s1 = (key0 + 16 <= row) ? sacc[1][r] : -1.0e30f;
            float t = fmaxf(s0, s1);
            t = fmaxf(t, __shfl_xor(t, 1));
            t = fmaxf(t, __shfl_xor(t, 2));
            t = fmaxf(t, __shfl_xor(t, 4));
            t = fmaxf(t, __shfl_xor(t, 8));
            const float mn   = fmaxf(m[r], t);
            const float corr = __expf(m[r] - mn);
            m[r] = mn;
            // f16-round p once; use the same value for l and for P@V
            const _Float16 p0h = (_Float16)__expf(s0 - mn);
            const _Float16 p1h = (_Float16)__expf(s1 - mn);
            float rs = (float)p0h + (float)p1h;
            rs += __shfl_xor(rs, 1);
            rs += __shfl_xor(rs, 2);
            rs += __shfl_xor(rs, 4);
            rs += __shfl_xor(rs, 8);
            l[r] = l[r] * corr + rs;
            #pragma unroll
            for (int gg = 0; gg < 4; ++gg) oacc[gg][r] *= corr;
            const int rowl = r + ((lane >> 4) << 3);
            Pt[rowl * LDS_STRIDE + rsel]      = p0h;
            Pt[rowl * LDS_STRIDE + 16 + rsel] = p1h;
        }
        asm volatile("s_wait_dscnt 0x0" ::: "memory");

        // ---- P fragment (A layout) from LDS
        h8 plo = *(const h8*)&Pt[rsel * LDS_STRIDE + kb];
        h8 phi = *(const h8*)&Pt[rsel * LDS_STRIDE + kb + 16];
        v16h pf = __builtin_shufflevector(plo, phi,
                    0,1,2,3,4,5,6,7,8,9,10,11,12,13,14,15);

        // ---- O += P @ V (V as B fragments: col = dim, k = key)
        #pragma unroll
        for (int gg = 0; gg < 4; ++gg) {
            const float* vb = qkv + (long)(b * T + j0) * C3 + 2 * C
                              + hq * 64 + gg * 16 + rsel;
            v16h vf;
            #pragma unroll
            for (int hh = 0; hh < 16; ++hh) {
                int kl = kb + (hh & 7) + ((hh >> 3) << 4);
                vf[hh] = (_Float16)vb[(long)kl * C3];
            }
            oacc[gg] = __builtin_amdgcn_wmma_f32_16x16x32_f16(
                false, pf, false, vf, (short)0, oacc[gg], false, false);
        }
    }

    // ---- normalize and store
    float inv[8];
    #pragma unroll
    for (int r = 0; r < 8; ++r) inv[r] = 1.f / l[r];
    #pragma unroll
    for (int gg = 0; gg < 4; ++gg) {
        #pragma unroll
        for (int r = 0; r < 8; ++r) {
            const int rowl = r + ((lane >> 4) << 3);
            y[(long)(b * T + qbase + rowl) * C + hq * 64 + gg * 16 + rsel]
                = oacc[gg][r] * inv[r];
        }
    }
}

// ---------------------------------------------------------------------------
// Host-side orchestration (all launches on `stream`, graph-capture safe)
// ---------------------------------------------------------------------------
extern "C" void kernel_launch(void* const* d_in, const int* in_sizes, int n_in,
                              void* d_out, int out_size, void* d_ws, size_t ws_size,
                              hipStream_t stream) {
    const int*   idx   = (const int*)  d_in[0];
    const float* wte   = (const float*)d_in[1];
    const float* wpe   = (const float*)d_in[2];
    const float* ln1w  = (const float*)d_in[3];
    const float* ln1b  = (const float*)d_in[4];
    const float* attnw = (const float*)d_in[5];
    const float* attnb = (const float*)d_in[6];
    const float* projw = (const float*)d_in[7];
    const float* projb = (const float*)d_in[8];
    const float* ln2w  = (const float*)d_in[9];
    const float* ln2b  = (const float*)d_in[10];
    const float* fcw   = (const float*)d_in[11];
    const float* fcb   = (const float*)d_in[12];
    const float* fc2w  = (const float*)d_in[13];
    const float* fc2b  = (const float*)d_in[14];
    const float* lnfw  = (const float*)d_in[15];
    const float* lnfb  = (const float*)d_in[16];
    float* logits = (float*)d_out;

    const int Bn = 4, T = 1024, C = 768, L = 6, Hn = 12, V = 50257;
    const int M = Bn * T; // 4096

    // workspace carve-up (floats): x | h | qkv | y | fc_act  (~126 MB)
    float* x    = (float*)d_ws;
    float* hbuf = x    + (size_t)M * C;
    float* qkv  = hbuf + (size_t)M * C;
    float* ybuf = qkv  + (size_t)M * 3 * C;
    float* fca  = ybuf + (size_t)M * C;

    {
        long tot = (long)M * C;
        embed_k<<<dim3((unsigned)((tot + 255) / 256)), 256, 0, stream>>>(
            idx, wte, wpe, x, M, C, T);
    }

    const dim3 blk(256);
    const int mTiles = M / 64;
    const int attnBlocks = (Bn * Hn * (T / 16)) / 4;  // 4 waves per block
    for (int l = 0; l < L; ++l) {
        layernorm_k<<<M, blk, 0, stream>>>(x, ln1w + (size_t)l * C, ln1b + (size_t)l * C, hbuf, C);

        gemm_wmma<true, false, false, false>
            <<<dim3((3 * C + 255) / 256, mTiles), blk, 0, stream>>>(
            hbuf, attnw + (size_t)l * C * 3 * C, attnb + (size_t)l * 3 * C,
            nullptr, qkv, M, 3 * C, C);

        attn_wmma_k<<<attnBlocks, 128, 0, stream>>>(qkv, ybuf, Bn, Hn, T);

        gemm_wmma<true, false, true, false>
            <<<dim3((C + 255) / 256, mTiles), blk, 0, stream>>>(
            ybuf, projw + (size_t)l * C * C, projb + (size_t)l * C,
            x, x, M, C, C);

        layernorm_k<<<M, blk, 0, stream>>>(x, ln2w + (size_t)l * C, ln2b + (size_t)l * C, hbuf, C);

        gemm_wmma<true, true, false, false>
            <<<dim3((4 * C + 255) / 256, mTiles), blk, 0, stream>>>(
            hbuf, fcw + (size_t)l * C * 4 * C, fcb + (size_t)l * 4 * C,
            nullptr, fca, M, 4 * C, C);

        gemm_wmma<true, false, true, false>
            <<<dim3((C + 255) / 256, mTiles), blk, 0, stream>>>(
            fca, fc2w + (size_t)l * 4 * C * C, fc2b + (size_t)l * C,
            x, x, M, C, 4 * C);
    }

    layernorm_k<<<M, blk, 0, stream>>>(x, lnfw, lnfb, hbuf, C);

    gemm_wmma<false, false, false, true>
        <<<dim3((V + 255) / 256, mTiles), blk, 0, stream>>>(
        hbuf, wte, nullptr, nullptr, logits, M, V, C);
}
